// TinyGpt_33578054320388
// MI455X (gfx1250) — compile-verified, matrix operand order checked by reading, not run on hardware
//
#include <hip/hip_runtime.h>
#include <math.h>

typedef __attribute__((ext_vector_type(16))) __bf16 v16bf;
typedef __attribute__((ext_vector_type(8)))  float  v8f;

union ABf {                 // one WMMA 16x32 bf16 operand (8 VGPRs / 32 bytes per lane)
  v16bf v;
  uint4 q[2];
  __bf16 h[16];
};

// ---------------- constants ----------------
static constexpr int kB = 2, kS = 2048, kE = 1024, kH = 16, kHD = 64, kDEPTH = 6, kV = 32000;
static constexpr int kM = kB * kS;          // 4096 token rows
static constexpr int k3E = 3 * kE;          // 3072

// ---------------- helpers ----------------
__device__ __forceinline__ __bf16 f2bf(float f) {
  union { float f; unsigned u; } x; x.f = f;
  unsigned r = x.u + 0x7FFFu + ((x.u >> 16) & 1u);   // round-to-nearest-even
  unsigned short hs = (unsigned short)(r >> 16);
  __bf16 out; __builtin_memcpy(&out, &hs, 2);
  return out;
}
__device__ __forceinline__ unsigned pack_bf16x2(float lo, float hi) {
  union { float f; unsigned u; } a, b; a.f = lo; b.f = hi;
  unsigned ra = a.u + 0x7FFFu + ((a.u >> 16) & 1u);
  unsigned rb = b.u + 0x7FFFu + ((b.u >> 16) & 1u);
  return (ra >> 16) | (rb & 0xFFFF0000u);
}
__device__ __forceinline__ float gelu_exact(float v) {
  return 0.5f * v * (1.0f + erff(v * 0.70710678118654752f));
}

// ---------------- embedding ----------------
__global__ __launch_bounds__(256) void embed_kernel(
    const int* __restrict__ idx, const float* __restrict__ tok,
    const float* __restrict__ pos, float* __restrict__ x) {
  int row = blockIdx.x;                 // b*S + s
  int t = idx[row];
  int s = row & (kS - 1);
  const float* tr = tok + (size_t)t * kE;
  const float* pr = pos + (size_t)s * kE;
  float* xr = x + (size_t)row * kE;
  for (int j = threadIdx.x; j < kE; j += 256) xr[j] = tr[j] + pr[j];
}

// ---------------- layernorm (f32 in -> bf16 out) ----------------
__global__ __launch_bounds__(256) void layernorm_bf16(
    const float* __restrict__ x, const float* __restrict__ w,
    const float* __restrict__ bvec, __bf16* __restrict__ out) {
  __shared__ float red[256];
  const int tid = threadIdx.x;
  const int row = blockIdx.x;
  const float* xr = x + (size_t)row * kE;
  float v[4]; float s = 0.f;
  for (int j = 0; j < 4; ++j) { v[j] = xr[tid + j * 256]; s += v[j]; }
  red[tid] = s; __syncthreads();
  for (int off = 128; off > 0; off >>= 1) { if (tid < off) red[tid] += red[tid + off]; __syncthreads(); }
  float mu = red[0] * (1.0f / kE);
  __syncthreads();
  s = 0.f;
  for (int j = 0; j < 4; ++j) { float d = v[j] - mu; s += d * d; }
  red[tid] = s; __syncthreads();
  for (int off = 128; off > 0; off >>= 1) { if (tid < off) red[tid] += red[tid + off]; __syncthreads(); }
  float rstd = rsqrtf(red[0] * (1.0f / kE) + 1e-5f);
  __bf16* orow = out + (size_t)row * kE;
  for (int j = 0; j < 4; ++j) {
    int c = tid + j * 256;
    orow[c] = f2bf((v[j] - mu) * rstd * w[c] + bvec[c]);
  }
}

// ---------------- residual add ----------------
__global__ __launch_bounds__(256) void resid_add(float* __restrict__ x, const float* __restrict__ y) {
  size_t i = (size_t)blockIdx.x * 256 + threadIdx.x;
  x[i] += y[i];
}

// ---------------- tiled bf16 WMMA GEMM: C = A[M,K] * W[N,K]^T (+bias)(+epilogue) ----------------
enum { EP_BF16 = 0, EP_GELU_BF16 = 1, EP_F32 = 2, EP_RESADD = 3 };

template <int EP, bool HAS_BIAS>
__global__ __launch_bounds__(256) void gemm_bf16(
    const __bf16* __restrict__ A,   // [M,K] bf16
    const float*  __restrict__ W,   // [N,K] f32 (converted per-tile)
    const float*  __restrict__ bias,
    void*  __restrict__ Cout,       // bf16 or f32 [M,N]
    float* __restrict__ resid,      // EP_RESADD target [M,N] f32
    int M, int N, int K) {
  __shared__ __bf16 sA[128][32];    // 8 KB
  __shared__ __bf16 sB[128][32];    // 8 KB
  const int tid  = threadIdx.x;
  const int lane = tid & 31, l16 = lane & 15, half = lane >> 4;
  const int wave = tid >> 5;
  const int wm = wave >> 1;         // 0..3 : 32-row slab
  const int wn = wave & 1;          // 0..1 : 64-col slab
  const int m0 = blockIdx.x * 128;
  const int n0 = blockIdx.y * 128;

  const v8f vz = {0.f, 0.f, 0.f, 0.f, 0.f, 0.f, 0.f, 0.f};
  v8f acc[2][4];
  for (int mt = 0; mt < 2; ++mt) for (int nt = 0; nt < 4; ++nt) acc[mt][nt] = vz;

  for (int k0 = 0; k0 < K; k0 += 32) {
    // stage A: 128x32 bf16, 2 x uint4 per thread
    for (int t = 0; t < 2; ++t) {
      int slot = tid + t * 256;           // 0..511
      int r = slot >> 2, c8 = slot & 3;
      *(uint4*)&sA[r][c8 * 8] =
          *(const uint4*)(A + (size_t)(m0 + r) * K + k0 + c8 * 8);
    }
    // stage W: 128x32 f32 -> bf16, 4 x float4 per thread
    for (int t = 0; t < 4; ++t) {
      int slot = tid + t * 256;           // 0..1023
      int r = slot >> 3, c4 = slot & 7;
      float4 f = *(const float4*)(W + (size_t)(n0 + r) * K + k0 + c4 * 4);
      uint2 p; p.x = pack_bf16x2(f.x, f.y); p.y = pack_bf16x2(f.z, f.w);
      *(uint2*)&sB[r][c4 * 4] = p;
    }
    if (k0 + 32 < K) {                    // global_prefetch_b8 for next tiles
      __builtin_prefetch(A + (size_t)(m0 + (tid >> 1)) * K + k0 + 32, 0, 0);
      __builtin_prefetch(W + (size_t)(n0 + (tid >> 1)) * K + k0 + 32, 0, 0);
    }
    __syncthreads();

    ABf af[2], bf[4];
    const int cb = half * 8;
    for (int mt = 0; mt < 2; ++mt) {
      int r = wm * 32 + mt * 16 + l16;
      af[mt].q[0] = *(const uint4*)&sA[r][cb];
      af[mt].q[1] = *(const uint4*)&sA[r][cb + 16];
    }
    for (int nt = 0; nt < 4; ++nt) {
      int r = wn * 64 + nt * 16 + l16;
      bf[nt].q[0] = *(const uint4*)&sB[r][cb];
      bf[nt].q[1] = *(const uint4*)&sB[r][cb + 16];
    }
    for (int mt = 0; mt < 2; ++mt)
      for (int nt = 0; nt < 4; ++nt)
        acc[mt][nt] = __builtin_amdgcn_wmma_f32_16x16x32_bf16(
            false, af[mt].v, false, bf[nt].v, (short)0, acc[mt][nt], false, false);
    __syncthreads();
  }

  // epilogue: C tile layout — lane 0-15: M=r, lanes 16-31: M=r+8 ; N = l16
  for (int mt = 0; mt < 2; ++mt)
    for (int nt = 0; nt < 4; ++nt)
      for (int r = 0; r < 8; ++r) {
        int mr = m0 + wm * 32 + mt * 16 + r + half * 8;
        int nc = n0 + wn * 64 + nt * 16 + l16;
        float val = acc[mt][nt][r];
        if (HAS_BIAS) val += bias[nc];
        size_t o = (size_t)mr * N + nc;
        if (EP == EP_GELU_BF16) val = gelu_exact(val);
        if (EP == EP_BF16 || EP == EP_GELU_BF16) ((__bf16*)Cout)[o] = f2bf(val);
        else if (EP == EP_F32)                   ((float*)Cout)[o] = val;
        else                                     resid[o] = resid[o] + val;
      }
}

// ---------------- flash attention: one wave per (b, h, 16-query tile) ----------------
__global__ __launch_bounds__(32) void attn_flash(
    const __bf16* __restrict__ qkv,   // [B,S,3E] bf16, col = i*E + h*64 + d
    float* __restrict__ o) {          // [B,S,E] f32
  __shared__ __bf16 sP[16][32];       // probs tile (C-layout -> A-layout bounce)
  __shared__ __bf16 sV[32][80];       // V 32 keys x 64 (padded row stride, 16B aligned)
  const int lane = threadIdx.x, l16 = lane & 15, half = lane >> 4;
  const int b = blockIdx.x >> 4, h = blockIdx.x & 15;
  const int q0 = blockIdx.y * 16;

  const __bf16* base = qkv + (size_t)b * kS * k3E + h * kHD;
  const v8f vz = {0.f, 0.f, 0.f, 0.f, 0.f, 0.f, 0.f, 0.f};

  // Q fragments (16x64 split into two 16x32 A-frags)
  ABf aq[2];
  {
    const __bf16* qp = base + (size_t)(q0 + l16) * k3E;
    for (int kc = 0; kc < 2; ++kc) {
      int d = kc * 32 + half * 8;
      aq[kc].q[0] = *(const uint4*)(qp + d);
      aq[kc].q[1] = *(const uint4*)(qp + d + 16);
    }
  }

  v8f oacc[4]; for (int nt = 0; nt < 4; ++nt) oacc[nt] = vz;
  float mrow[8], lrow[8];
  for (int r = 0; r < 8; ++r) { mrow[r] = -INFINITY; lrow[r] = 0.f; }

  const int nkb = (q0 + 16 + 31) >> 5;
  for (int kb = 0; kb < nkb; ++kb) {
    const int kbase = kb * 32;
    // stage V: lane = key row, 64 bf16 each
    {
      const __bf16* vp = base + 2 * kE + (size_t)(kbase + lane) * k3E;
      for (int j = 0; j < 8; ++j)
        *(uint4*)&sV[lane][j * 8] = *(const uint4*)(vp + j * 8);
    }
    // scores S = Q K^T (two 16x16 column tiles)
    v8f sc[2];
    for (int nb = 0; nb < 2; ++nb) {
      sc[nb] = vz;
      const __bf16* kp = base + kE + (size_t)(kbase + nb * 16 + l16) * k3E;
      for (int kc = 0; kc < 2; ++kc) {
        ABf bk;
        int d = kc * 32 + half * 8;
        bk.q[0] = *(const uint4*)(kp + d);
        bk.q[1] = *(const uint4*)(kp + d + 16);
        sc[nb] = __builtin_amdgcn_wmma_f32_16x16x32_bf16(
            false, aq[kc].v, false, bk.v, (short)0, sc[nb], false, false);
      }
    }
    // scale + causal mask
    for (int nb = 0; nb < 2; ++nb) {
      int key = kbase + nb * 16 + l16;
      for (int r = 0; r < 8; ++r) {
        int q = q0 + r + half * 8;
        float v = sc[nb][r] * 0.125f;           // 1/sqrt(64)
        sc[nb][r] = (key <= q) ? v : -INFINITY;
      }
    }
    // online softmax stats (rows live across the 16-lane half-groups)
    float alpha[8];
    for (int r = 0; r < 8; ++r) {
      float mx = fmaxf(sc[0][r], sc[1][r]);
      for (int off = 1; off < 16; off <<= 1) mx = fmaxf(mx, __shfl_xor(mx, off, 32));
      float mnew = fmaxf(mrow[r], mx);
      alpha[r] = expf(mrow[r] - mnew);
      mrow[r] = mnew;
      float rs = 0.f;
      for (int nb = 0; nb < 2; ++nb) {
        float p = expf(sc[nb][r] - mnew);
        sc[nb][r] = p; rs += p;
      }
      for (int off = 1; off < 16; off <<= 1) rs += __shfl_xor(rs, off, 32);
      lrow[r] = lrow[r] * alpha[r] + rs;
    }
    // rescale O, bounce P through LDS (C-layout -> A-layout); same-wave DS ops are in-order
    for (int nt = 0; nt < 4; ++nt)
      for (int r = 0; r < 8; ++r) oacc[nt][r] *= alpha[r];
    for (int nb = 0; nb < 2; ++nb)
      for (int r = 0; r < 8; ++r)
        sP[r + half * 8][nb * 16 + l16] = f2bf(sc[nb][r]);
    ABf ap;
    ap.q[0] = *(const uint4*)&sP[l16][half * 8];
    ap.q[1] = *(const uint4*)&sP[l16][half * 8 + 16];
    // O += P V  (B-frag: lane = output column d, elements j, 8+j = keys 8*half+j, 16+8*half+j)
    for (int nt = 0; nt < 4; ++nt) {
      ABf bv;
      int d = nt * 16 + l16;
      for (int j = 0; j < 8; ++j) {
        bv.h[j]     = sV[half * 8 + j][d];
        bv.h[8 + j] = sV[16 + half * 8 + j][d];
      }
      oacc[nt] = __builtin_amdgcn_wmma_f32_16x16x32_bf16(
          false, ap.v, false, bv.v, (short)0, oacc[nt], false, false);
    }
  }
  // write O / l
  for (int nt = 0; nt < 4; ++nt)
    for (int r = 0; r < 8; ++r) {
      int q = q0 + r + half * 8;
      int col = h * kHD + nt * 16 + l16;
      o[((size_t)(b * kS + q)) * kE + col] = oacc[nt][r] / lrow[r];
    }
}

// ---------------- driver ----------------
extern "C" void kernel_launch(void* const* d_in, const int* in_sizes, int n_in,
                              void* d_out, int out_size, void* d_ws, size_t ws_size,
                              hipStream_t stream) {
  const int*   idx    = (const int*)  d_in[0];
  const float* tok    = (const float*)d_in[1];
  const float* pos    = (const float*)d_in[2];
  const float* qkv_w  = (const float*)d_in[3];
  const float* qkv_b  = (const float*)d_in[4];
  const float* ln1_w  = (const float*)d_in[5];
  const float* ln1_b  = (const float*)d_in[6];
  const float* ln2_w  = (const float*)d_in[7];
  const float* ln2_b  = (const float*)d_in[8];
  const float* ff1_w  = (const float*)d_in[9];
  const float* ff1_b  = (const float*)d_in[10];
  const float* ff2_w  = (const float*)d_in[11];
  const float* ff2_b  = (const float*)d_in[12];
  const float* fin_w  = (const float*)d_in[13];
  const float* fin_b  = (const float*)d_in[14];
  const float* proj_w = (const float*)d_in[15];
  (void)in_sizes; (void)n_in; (void)out_size; (void)ws_size;

  char* ws = (char*)d_ws;
  float*  x    = (float*) (ws);                                   // 4096*1024 f32  (16 MB)
  __bf16* xn   = (__bf16*)(ws + (size_t)16777216);                // 4096*1024 bf16 ( 8 MB)
  __bf16* qkvb = (__bf16*)(ws + (size_t)25165824);                // 4096*3072 bf16 (24 MB)
  float*  attn = (float*) (ws + (size_t)50331648);                // 4096*1024 f32  (16 MB)
  __bf16* h1   = (__bf16*)(ws + (size_t)67108864);                // 4096*4096 bf16 (32 MB)

  embed_kernel<<<kM, 256, 0, stream>>>(idx, tok, pos, x);

  for (int l = 0; l < kDEPTH; ++l) {
    layernorm_bf16<<<kM, 256, 0, stream>>>(x, ln1_w + l * kE, ln1_b + l * kE, xn);
    gemm_bf16<EP_BF16, true><<<dim3(kM / 128, k3E / 128), 256, 0, stream>>>(
        xn, qkv_w + (size_t)l * k3E * kE, qkv_b + (size_t)l * k3E,
        (void*)qkvb, nullptr, kM, k3E, kE);
    attn_flash<<<dim3(kB * kH, kS / 16), 32, 0, stream>>>(qkvb, attn);
    resid_add<<<(kM * kE) / 256, 256, 0, stream>>>(x, attn);
    layernorm_bf16<<<kM, 256, 0, stream>>>(x, ln2_w + l * kE, ln2_b + l * kE, xn);
    gemm_bf16<EP_GELU_BF16, true><<<dim3(kM / 128, (4 * kE) / 128), 256, 0, stream>>>(
        xn, ff1_w + (size_t)l * 4 * kE * kE, ff1_b + (size_t)l * 4 * kE,
        (void*)h1, nullptr, kM, 4 * kE, kE);
    gemm_bf16<EP_RESADD, true><<<dim3(kM / 128, kE / 128), 256, 0, stream>>>(
        h1, ff2_w + (size_t)l * kE * 4 * kE, ff2_b + (size_t)l * kE,
        nullptr, x, kM, kE, 4 * kE);
  }

  layernorm_bf16<<<kM, 256, 0, stream>>>(x, fin_w, fin_b, xn);
  gemm_bf16<EP_F32, false><<<dim3(kM / 128, kV / 128), 256, 0, stream>>>(
      xn, proj_w, nullptr, d_out, nullptr, kM, kV, kE);
}
